// FastAttention_56169582297260
// MI455X (gfx1250) — matrix operand order repaired and checked by reading, not compiled, not run on hardware
//
#include <hip/hip_runtime.h>
#include <stdint.h>

// ---------------------------------------------------------------------------
// Causal MHA block for MI455X (gfx1250, wave32, WMMA + Tensor Data Mover).
//   cvt:  x, W_attn^T, W_proj^T  -> bf16 (ws)           (~75MB @23.3TB/s ~ 3us)
//   GEMM1 qkv = xb @ Wat^T + b   (TDM double-buffered, 16 wmma / K-step-64)
//   flash attention per (b,h)    -> y (bf16 ws)
//   GEMM2 out = y @ Wpt^T + b    (same TDM GEMM, f32 out)
// ~137 GFLOP, ~50MB inputs => compute bound => everything on
// v_wmma_f32_16x16x32_bf16; data movement on TDM (TENSORcnt) so VALU/VMEM
// issue slots stay free for WMMA + fragment ds_load_b128.
// ---------------------------------------------------------------------------

#define BATCH 4
#define SEQ   2048
#define CH    1024
#define NHEAD 16
#define HDIM  64
#define KDIM  1024            // contraction dim of both GEMMs (= CH)

typedef __attribute__((ext_vector_type(16))) __bf16 v16bf;
typedef __attribute__((ext_vector_type(8)))  __bf16 v8bf;
typedef __attribute__((ext_vector_type(8)))  float  v8f;
typedef __attribute__((ext_vector_type(4)))  unsigned int v4u;
typedef __attribute__((ext_vector_type(8)))  int    v8i;
typedef __attribute__((ext_vector_type(4)))  int    v4i;

static __device__ inline v16bf join16(v8bf lo, v8bf hi) {
  return __builtin_shufflevector(lo, hi, 0,1,2,3,4,5,6,7,8,9,10,11,12,13,14,15);
}
static __device__ inline v8f zero8() {
  v8f z;
#pragma unroll
  for (int i = 0; i < 8; ++i) z[i] = 0.0f;
  return z;
}
static __device__ inline v8f wmma_bf16(v16bf a, v16bf b, v8f c) {
  return __builtin_amdgcn_wmma_f32_16x16x32_bf16(false, a, false, b, (short)0, c,
                                                 false, false);
}

// ---------------------------------------------------------------------------
// TDM: DMA a 128-row x 64-element (bf16) 2D tile, row stride KDIM elements,
// into LDS with 16B pad per 128B row => LDS row stride 144B (36 DW): rows hit
// disjoint 4-bank spans mod 64 => conflict-free b128 fragment loads.
// D# group0: count=1 | lds_addr | global_addr | type=2.
// D# group1: data_size=2B, pad_enable, pad_interval=32DW, pad_amount=4DW,
//            tensor_dim0=KDIM, tensor_dim1=1M, tile=64x128, dim0_stride=KDIM.
// ---------------------------------------------------------------------------
static __device__ inline void tdm_load_tile_128x64(const __bf16* gsrc,
                                                   unsigned lds_off) {
  unsigned long long ga = (unsigned long long)(uintptr_t)gsrc;
  v4u g0;
  g0[0] = 1u;                                           // count=1 (valid)
  g0[1] = lds_off;                                      // LDS byte address
  g0[2] = (unsigned)ga;                                 // global_addr[31:0]
  g0[3] = (unsigned)((ga >> 32) & 0x01FFFFFFu) | (2u << 30);  // addr[56:32]|type=2
  v8i g1;
  g1[0] = (1 << 16) | (1 << 20) | (4 << 22) | (3 << 25);  // 2B | pad 4DW/32DW
  g1[1] = (int)((KDIM & 0xFFFF) << 16);                 // tensor_dim0 lo16
  g1[2] = (KDIM >> 16);                                 // dim0 hi | dim1 lo16(0)
  g1[3] = 16 | (64 << 16);                              // dim1 hi (1<<20) | tile_dim0=64
  g1[4] = 128;                                          // tile_dim1=128, tile_dim2=0
  g1[5] = KDIM;                                         // tensor_dim0_stride lo32
  g1[6] = 0;
  g1[7] = 0;
  v4i z4 = {0, 0, 0, 0};
#if __has_include(<hip/amd_detail/amd_gfx1250_TDM.h>)
  v8i z8 = {0, 0, 0, 0, 0, 0, 0, 0};                    // therock: 6-arg form
  __builtin_amdgcn_tensor_load_to_lds(g0, g1, z4, z4, z8, 0);
#else
  __builtin_amdgcn_tensor_load_to_lds(g0, g1, z4, z4, 0);  // ROCm 7.2: 5-arg
#endif
}

// ---------------------------------------------------------------------------
// Elementwise f32 -> bf16 (float4 per thread).
// ---------------------------------------------------------------------------
__global__ __launch_bounds__(256) void cvt_bf16(const float* __restrict__ in,
                                                __bf16* __restrict__ out) {
  size_t i = (size_t)blockIdx.x * 256 + threadIdx.x;
  float4 v = reinterpret_cast<const float4*>(in)[i];
  out[i * 4 + 0] = (__bf16)v.x;
  out[i * 4 + 1] = (__bf16)v.y;
  out[i * 4 + 2] = (__bf16)v.z;
  out[i * 4 + 3] = (__bf16)v.w;
}

// in[K][N] f32  ->  out[N][K] bf16 (transposed, coalesced reads).
__global__ __launch_bounds__(256) void cvt_transpose_bf16(
    const float* __restrict__ in, __bf16* __restrict__ out, int K, int N) {
  int n = blockIdx.x * 256 + threadIdx.x;
  int k = blockIdx.y;
  out[(size_t)n * K + k] = (__bf16)in[(size_t)k * N + n];
}

// ---------------------------------------------------------------------------
// GEMM: out[M,N] = A[M,K] * Bt[N,K]^T + bias[N], A/Bt bf16, K = KDIM.
// 256 threads = 8 wave32s (4x2), block tile 128x128, wave tile 32x64.
// K-step 64: TDM double-buffers both 128x64 tiles; wave0 issues the next
// buffer's DMA, waits tensorcnt<=2 (current buffer done), barrier publishes.
// Fragments: contiguous 16B ds_load_b128 per the CDNA5 16-bit layouts
// (A/Bt rows: lane<16 K{0..7,16..23}, lane>=16 K{8..15,24..31}).
// ---------------------------------------------------------------------------
template <typename OT>
__global__ __launch_bounds__(256) void gemm_tn_wmma_tdm(
    const __bf16* __restrict__ A, const __bf16* __restrict__ Bt,
    const float* __restrict__ bias, OT* __restrict__ out, int M, int N) {
  __shared__ alignas(16) __bf16 As[2][128][72];   // 144B rows (TDM pad)
  __shared__ alignas(16) __bf16 Bs[2][128][72];

  const int tid  = threadIdx.x;
  const int lane = tid & 31, w = tid >> 5;
  const int ln   = lane & 15, half = lane >> 4;
  const int wm   = w & 3, wn = w >> 2;
  const int m0   = blockIdx.x * 128, n0 = blockIdx.y * 128;

  v8f acc[2][4];
#pragma unroll
  for (int mi = 0; mi < 2; ++mi)
#pragma unroll
    for (int ni = 0; ni < 4; ++ni) acc[mi][ni] = zero8();

  if (w == 0) {                              // wave-uniform: prologue DMA
    tdm_load_tile_128x64(A + (size_t)m0 * KDIM,
                         (unsigned)(uintptr_t)&As[0][0][0]);
    tdm_load_tile_128x64(Bt + (size_t)n0 * KDIM,
                         (unsigned)(uintptr_t)&Bs[0][0][0]);
  }

  for (int k0 = 0; k0 < KDIM; k0 += 64) {
    const int cur = (k0 >> 6) & 1;
    if (w == 0 && k0 + 64 < KDIM) {          // issue next buffer, then wait cur
      tdm_load_tile_128x64(A + (size_t)m0 * KDIM + k0 + 64,
                           (unsigned)(uintptr_t)&As[cur ^ 1][0][0]);
      tdm_load_tile_128x64(Bt + (size_t)n0 * KDIM + k0 + 64,
                           (unsigned)(uintptr_t)&Bs[cur ^ 1][0][0]);
      __builtin_amdgcn_s_wait_tensorcnt(2);
    } else {
      __builtin_amdgcn_s_wait_tensorcnt(0);  // no-op for waves 1..7
    }
    __syncthreads();

#pragma unroll
    for (int ks = 0; ks < 2; ++ks) {
      v16bf af[2];
#pragma unroll
      for (int mi = 0; mi < 2; ++mi) {
        const __bf16* p = &As[cur][wm * 32 + mi * 16 + ln][ks * 32 + half * 8];
        af[mi] = join16(*(const v8bf*)p, *(const v8bf*)(p + 16));
      }
#pragma unroll
      for (int ni = 0; ni < 4; ++ni) {
        const __bf16* p = &Bs[cur][wn * 64 + ni * 16 + ln][ks * 32 + half * 8];
        v16bf bf = join16(*(const v8bf*)p, *(const v8bf*)(p + 16));
#pragma unroll
        for (int mi = 0; mi < 2; ++mi)
          acc[mi][ni] = wmma_bf16(af[mi], bf, acc[mi][ni]);
      }
    }
    __syncthreads();
  }

  // Epilogue: C layout row = r + 8*half, col = ln; 16-lane halves write
  // contiguous 32B (bf16) / 64B (f32) spans => coalesced b16/b32 stores.
#pragma unroll
  for (int mi = 0; mi < 2; ++mi)
#pragma unroll
    for (int ni = 0; ni < 4; ++ni) {
      int col = n0 + wn * 64 + ni * 16 + ln;
      float bv = bias[col];
#pragma unroll
      for (int r = 0; r < 8; ++r) {
        int row = m0 + wm * 32 + mi * 16 + r + 8 * half;
        out[(size_t)row * N + col] = (OT)(acc[mi][ni][r] + bv);
      }
    }
}

// ---------------------------------------------------------------------------
// Flash attention (unchanged from prior round — already tight WMMA stream).
// One block per (b, h, 128-query tile); 8 waves x 16 rows. Q fragments loop
// invariant; K^T fragments straight from row-major qkv; V staged transposed;
// online softmax on the C layout via __shfl_xor(1,2,4,8) within wave32 halves.
// ---------------------------------------------------------------------------
__global__ __launch_bounds__(256) void flash_attn_wmma(
    const __bf16* __restrict__ qkv, __bf16* __restrict__ y) {
  __shared__ alignas(16) __bf16 Vt[HDIM][40];    // transposed V: [d][key]
  __shared__ alignas(16) __bf16 Ps[8][16][40];   // per-wave P tile 16x32

  const int tid  = threadIdx.x;
  const int lane = tid & 31, w = tid >> 5;
  const int ln   = lane & 15, half = lane >> 4;
  const int q0 = blockIdx.x * 128;
  const int h  = blockIdx.y;
  const int b  = blockIdx.z;
  const int qr0 = q0 + w * 16;
  const int qmax = qr0 + 15;                      // wave-uniform guard
  const size_t rs = 3 * CH;
  const __bf16* qb = qkv + (size_t)b * SEQ * rs;

  v16bf qa[2];
#pragma unroll
  for (int s = 0; s < 2; ++s) {
    const __bf16* p = qb + (size_t)(qr0 + ln) * rs + h * HDIM + s * 32 + half * 8;
    qa[s] = join16(*(const v8bf*)p, *(const v8bf*)(p + 16));
  }

  v8f oacc[4];
#pragma unroll
  for (int t = 0; t < 4; ++t) oacc[t] = zero8();
  float m[8], l[8];
#pragma unroll
  for (int r = 0; r < 8; ++r) { m[r] = -1e30f; l[r] = 0.0f; }

  const int kend = q0 + 128;
  for (int kb = 0; kb < kend; kb += 32) {
#pragma unroll
    for (int it = 0; it < 8; ++it) {
      int e = tid + it * 256;
      int key = e >> 6, dc = e & 63;
      Vt[dc][key] = qb[(size_t)(kb + key) * rs + 2 * CH + h * HDIM + dc];
    }
    __syncthreads();

    if (kb <= qmax) {
      v8f sacc[2] = {zero8(), zero8()};
#pragma unroll
      for (int s = 0; s < 2; ++s)
#pragma unroll
        for (int ti = 0; ti < 2; ++ti) {
          const __bf16* p = qb + (size_t)(kb + ti * 16 + ln) * rs + CH +
                            h * HDIM + s * 32 + half * 8;
          v16bf kf = join16(*(const v8bf*)p, *(const v8bf*)(p + 16));
          sacc[ti] = wmma_bf16(qa[s], kf, sacc[ti]);
        }

      float sm[2][8];
#pragma unroll
      for (int ti = 0; ti < 2; ++ti)
#pragma unroll
        for (int r = 0; r < 8; ++r) {
          int qrow = qr0 + r + 8 * half;
          int key  = kb + ti * 16 + ln;
          float sv = sacc[ti][r] * 0.125f;
          sm[ti][r] = (key <= qrow) ? sv : -1e30f;
        }

      float alpha[8];
#pragma unroll
      for (int r = 0; r < 8; ++r) {
        float rm = fmaxf(sm[0][r], sm[1][r]);
#pragma unroll
        for (int mk = 1; mk < 16; mk <<= 1) rm = fmaxf(rm, __shfl_xor(rm, mk, 32));
        float mn = fmaxf(m[r], rm);
        alpha[r] = __expf(m[r] - mn);
        m[r] = mn;
      }
#pragma unroll
      for (int ti = 0; ti < 2; ++ti)
#pragma unroll
        for (int r = 0; r < 8; ++r) {
          float p = __expf(sm[ti][r] - m[r]);
          sm[ti][r] = p;
          Ps[w][r + 8 * half][ti * 16 + ln] = (__bf16)p;
        }
#pragma unroll
      for (int r = 0; r < 8; ++r) {
        float rsum = sm[0][r] + sm[1][r];
#pragma unroll
        for (int mk = 1; mk < 16; mk <<= 1) rsum += __shfl_xor(rsum, mk, 32);
        l[r] = l[r] * alpha[r] + rsum;
#pragma unroll
        for (int t = 0; t < 4; ++t) oacc[t][r] *= alpha[r];
      }

      const __bf16* pp = &Ps[w][ln][half * 8];
      v16bf pf = join16(*(const v8bf*)pp, *(const v8bf*)(pp + 16));
#pragma unroll
      for (int t = 0; t < 4; ++t) {
        const __bf16* vp = &Vt[t * 16 + ln][half * 8];
        v16bf vf = join16(*(const v8bf*)vp, *(const v8bf*)(vp + 16));
        oacc[t] = wmma_bf16(pf, vf, oacc[t]);
      }
    }
    __syncthreads();
  }

  __bf16* yb = y + (size_t)b * SEQ * CH;
#pragma unroll
  for (int r = 0; r < 8; ++r) {
    float inv = 1.0f / l[r];
    int row = qr0 + r + 8 * half;
#pragma unroll
    for (int t = 0; t < 4; ++t) {
      int col = h * HDIM + t * 16 + ln;
      yb[(size_t)row * CH + col] = (__bf16)(oacc[t][r] * inv);
    }
  }
}

// ---------------------------------------------------------------------------
extern "C" void kernel_launch(void* const* d_in, const int* in_sizes, int n_in,
                              void* d_out, int out_size, void* d_ws,
                              size_t ws_size, hipStream_t stream) {
  const float* x      = (const float*)d_in[0];   // [4,2048,1024]
  const float* W_attn = (const float*)d_in[1];   // [1024,3072]
  const float* b_attn = (const float*)d_in[2];   // [3072]
  const float* W_proj = (const float*)d_in[3];   // [1024,1024]
  const float* b_proj = (const float*)d_in[4];   // [1024]
  float* out = (float*)d_out;

  const int M = BATCH * SEQ;                     // 8192
  __bf16* xb  = (__bf16*)d_ws;                   // 8192x1024
  __bf16* Wat = xb  + (size_t)M * CH;            // 3072x1024 (W_attn^T)
  __bf16* Wpt = Wat + (size_t)3 * CH * CH;       // 1024x1024 (W_proj^T)
  __bf16* qkv = Wpt + (size_t)CH * CH;           // 8192x3072
  __bf16* yws = qkv + (size_t)M * 3 * CH;        // 8192x1024   (total 88MB)

  // 0) bf16 operand prep (x straight, weights transposed for TN GEMM).
  cvt_bf16<<<(M * CH) / 1024, 256, 0, stream>>>(x, xb);
  cvt_transpose_bf16<<<dim3((3 * CH) / 256, CH), 256, 0, stream>>>(
      W_attn, Wat, CH, 3 * CH);
  cvt_transpose_bf16<<<dim3(CH / 256, CH), 256, 0, stream>>>(
      W_proj, Wpt, CH, CH);

  // 1) qkv = xb @ Wat^T + b_attn (bf16 out)
  gemm_tn_wmma_tdm<__bf16><<<dim3(M / 128, (3 * CH) / 128), 256, 0, stream>>>(
      xb, Wat, b_attn, qkv, M, 3 * CH);

  // 2) flash attention -> y (bf16)
  flash_attn_wmma<<<dim3(SEQ / 128, NHEAD, BATCH), 256, 0, stream>>>(qkv, yws);

  // 3) out = y @ Wpt^T + b_proj (f32 out)
  gemm_tn_wmma_tdm<float><<<dim3(M / 128, CH / 128), 256, 0, stream>>>(
      yws, Wpt, b_proj, out, M, CH);
}
